// SCX_Block_22539988369668
// MI455X (gfx1250) — compile-verified, matrix operand order checked by reading, not run on hardware
//
#include <hip/hip_runtime.h>

// ---------------- CDNA5 types ----------------
typedef __bf16 bf16;
typedef __attribute__((ext_vector_type(16))) bf16  v16bf;
typedef __attribute__((ext_vector_type(8)))  float v8f;
typedef __attribute__((ext_vector_type(4)))  bf16  v4bf;

// ---------------- problem constants ----------------
#define BS_     384            // B * SEG_NUM
#define SEGN_   24
#define GRP_    10
#define HDS_    8
#define DM_     256
#define VARN_   862
#define KSEL_   87
#define DH_     32
#define M1_     (BS_ * VARN_)  // 331008 rows of the big GEMMs (divisible by 128)

// ---- gfx1250 async copy: global -> LDS, 16 bytes per lane, tracked by ASYNCcnt ----
__device__ __forceinline__ void async_ld16(unsigned lds_addr, const void* gptr) {
    asm volatile("global_load_async_to_lds_b128 %0, %1, off"
                 :: "v"(lds_addr), "v"(gptr)
                 : "memory");
}
__device__ __forceinline__ void wait_async0() {
    asm volatile("s_wait_asynccnt 0x0" ::: "memory");
}
__device__ __forceinline__ unsigned lds_off(const void* p) {
    // flat LDS aperture: addr[31:0] is the wave-relative LDS byte offset (ISA 10.2)
    return (unsigned)(unsigned long long)p;
}

// Load a 16-element bf16 WMMA fragment (A or B) for one lane from a 32-wide row.
// ISA 16-bit A/B layout: element e <-> K = (e<8?0:16) + 8*half + (e&7)
// => elements 0..7 come from row[8*half .. +7], 8..15 from row[16+8*half .. +7].
__device__ __forceinline__ v16bf load_frag32(const bf16* __restrict__ row, int half) {
    union { uint4 u[2]; v16bf v; } U;
    U.u[0] = *(const uint4*)(row + 8 * half);
    U.u[1] = *(const uint4*)(row + 16 + 8 * half);
    return U.v;
}

// ---------------- K0a: f32 -> bf16 weight cast with transpose (WT[n][k] = W[k][n]) --
__global__ void cast_wT_kernel(const float* __restrict__ w, bf16* __restrict__ o) {
    int t = blockIdx.x * 256 + threadIdx.x;       // 0..65535
    int k = t >> 8, n = t & 255;
    o[n * DM_ + k] = (bf16)w[t];                  // w[t] = w[k*256+n], coalesced read
}

// ---------------- K0b: q = cluster @ Wq + bq, rearranged to [seg][h][g][32] bf16 ----
__global__ void compute_q_kernel(const float* __restrict__ cl, const float* __restrict__ Wq,
                                 const float* __restrict__ bq, bf16* __restrict__ qb) {
    int t = blockIdx.x * 256 + threadIdx.x;       // 0 .. 24*10*256-1
    if (t >= SEGN_ * GRP_ * DM_) return;
    int c = t & 255;
    int sg = t >> 8;
    int g = sg % GRP_, seg = sg / GRP_;
    const float* row = cl + (size_t)(seg * GRP_ + g) * DM_;
    float acc = bq[c];
    for (int k = 0; k < DM_; ++k) acc += row[k] * Wq[k * DM_ + c];
    int h = c >> 5, d = c & 31;
    qb[(((size_t)seg * HDS_ + h) * GRP_ + g) * DH_ + d] = (bf16)acc;
}

// ---------------- K1: y = log(relu(x)+1) cast to bf16 (float4 -> 4x bf16) --------
__global__ void act_cast_kernel(const float4* __restrict__ x, v4bf* __restrict__ y, int nv) {
    int i = blockIdx.x * 256 + threadIdx.x;
    if (i >= nv) return;
    float4 t = x[i];
    v4bf o;
    o[0] = (bf16)__logf(fmaxf(t.x, 0.f) + 1.f);
    o[1] = (bf16)__logf(fmaxf(t.y, 0.f) + 1.f);
    o[2] = (bf16)__logf(fmaxf(t.z, 0.f) + 1.f);
    o[3] = (bf16)__logf(fmaxf(t.w, 0.f) + 1.f);
    y[i] = o;
}

// ---------------- K2/K6: tiled bf16 WMMA GEMM, C[M,256] = A[M,256] @ BT^T + bias
// BT is the transposed weight, row-major [n][k].  block = 256 threads = 8 waves;
// tile 128(M) x 128(N); KB = 32; double-buffered async-to-LDS staging.
// gridDim.z selects (BT0,bias0,C0) vs (BT1,bias1,C1) so k/v share A traffic in L2.
template <bool OUTF32>
__global__ __launch_bounds__(256) void gemm_bf16_256(const bf16* __restrict__ A,
                                                     const bf16* __restrict__ BT0,
                                                     const bf16* __restrict__ BT1,
                                                     const float* __restrict__ bias0,
                                                     const float* __restrict__ bias1,
                                                     void* __restrict__ C0,
                                                     void* __restrict__ C1) {
    __shared__ __align__(16) bf16 As[2][128 * 32];   // [m][k]
    __shared__ __align__(16) bf16 Bs[2][128 * 32];   // [n][k]

    const bf16*  BT   = blockIdx.z ? BT1 : BT0;
    const float* bias = blockIdx.z ? bias1 : bias0;
    void*        Cout = blockIdx.z ? C1 : C0;

    const int tid = threadIdx.x;
    const int mBase = blockIdx.x * 128;
    const int nBase = blockIdx.y * 128;
    const int wave = tid >> 5, lane = tid & 31;
    const int wm = wave >> 2, wn = wave & 3;      // 2 waves along M, 4 along N
    const int half = lane >> 4, mr = lane & 15;

    auto stage = [&](int buf, int kk) {
#pragma unroll
        for (int it = 0; it < 2; ++it) {
            int cc = tid + 256 * it;              // 0..511
            int m = cc >> 2, ko = (cc & 3) * 8;
            async_ld16(lds_off(&As[buf][m * 32 + ko]),
                       &A[(size_t)(mBase + m) * DM_ + kk + ko]);
        }
#pragma unroll
        for (int it = 0; it < 2; ++it) {
            int cc = tid + 256 * it;
            int m = cc >> 2, ko = (cc & 3) * 8;
            async_ld16(lds_off(&Bs[buf][m * 32 + ko]),
                       &BT[(size_t)(nBase + m) * DM_ + kk + ko]);
        }
    };

    v8f zero8 = {};
    v8f acc[4][2];
#pragma unroll
    for (int i = 0; i < 4; ++i)
#pragma unroll
        for (int j = 0; j < 2; ++j) acc[i][j] = zero8;

    stage(0, 0);
    wait_async0();
    __syncthreads();

    for (int ki = 0; ki < 8; ++ki) {              // 8 K-steps of 32
        const int cur = ki & 1;
        if (ki < 7) stage(cur ^ 1, (ki + 1) * 32);

        v16bf af[4], bfr[2];
#pragma unroll
        for (int i = 0; i < 4; ++i)
            af[i] = load_frag32(&As[cur][(wm * 64 + i * 16 + mr) * 32], half);
#pragma unroll
        for (int j = 0; j < 2; ++j)
            bfr[j] = load_frag32(&Bs[cur][(wn * 32 + j * 16 + mr) * 32], half);

#pragma unroll
        for (int i = 0; i < 4; ++i)
#pragma unroll
            for (int j = 0; j < 2; ++j)
                acc[i][j] = __builtin_amdgcn_wmma_f32_16x16x32_bf16(
                    false, af[i], false, bfr[j], (short)0, acc[i][j], false, false);

        if (ki < 7) wait_async0();                // own async copies landed in LDS
        __syncthreads();                          // everyone done reading/writing
    }

    // epilogue: D layout: vgpr r -> m = r + 8*half, lane%16 -> n
#pragma unroll
    for (int i = 0; i < 4; ++i)
#pragma unroll
        for (int j = 0; j < 2; ++j) {
            int col = nBase + wn * 32 + j * 16 + mr;
            float bval = bias[col];
#pragma unroll
            for (int r = 0; r < 8; ++r) {
                int row = mBase + wm * 64 + i * 16 + r + 8 * half;
                float v = acc[i][j][r] + bval;
                if (OUTF32)
                    ((float*)Cout)[(size_t)row * DM_ + col] = v;
                else
                    ((bf16*)Cout)[(size_t)row * DM_ + col] = (bf16)v;
            }
        }
}

// ---------------- K3: scores[s,h,g,n] = q·k / sqrt(32) via single-K-step WMMA -----
__global__ __launch_bounds__(256) void scores_kernel(const bf16* __restrict__ qb,
                                                     const bf16* __restrict__ kb,
                                                     float* __restrict__ scores) {
    const int sh = blockIdx.x;                 // 0..3071
    const int s = sh >> 3, h = sh & 7;
    const int seg = s % SEGN_;
    const int wave = threadIdx.x >> 5, lane = threadIdx.x & 31;
    const int half = lane >> 4, mr = lane & 15;

    // A fragment: q rows (g = mr, pad g>=10 with zero)
    v16bf a = {};
    if (mr < GRP_)
        a = load_frag32(qb + (((size_t)seg * HDS_ + h) * GRP_ + mr) * DH_, half);

    float* sbase = scores + ((size_t)s * HDS_ + h) * GRP_ * VARN_;
    const float scale = 0.17677669529663687f;  // 1/sqrt(32)

    for (int tile = wave; tile < 54; tile += 8) {   // 54*16 = 864 >= 862
        int n = tile * 16 + mr;
        v16bf b = {};
        if (n < VARN_)
            b = load_frag32(kb + ((size_t)(s * VARN_ + n)) * DM_ + h * DH_, half);
        v8f c = {};
        c = __builtin_amdgcn_wmma_f32_16x16x32_bf16(false, a, false, b, (short)0, c,
                                                    false, false);
#pragma unroll
        for (int r = 0; r < 8; ++r) {
            int g = r + 8 * half;
            if (g < GRP_ && n < VARN_)
                sbase[(size_t)g * VARN_ + n] = c[r] * scale;
        }
    }
}

// ---------------- K4: top-87 + softmax per (s,h,g) row ----------------
__global__ __launch_bounds__(256) void topk_kernel(const float* __restrict__ scores,
                                                   float* __restrict__ attn_out,
                                                   int* __restrict__ idx_out) {
    const int row = blockIdx.x;                // 0..30719
    const float* src = scores + (size_t)row * VARN_;
    __shared__ float vals[VARN_];
    __shared__ float rv[256];
    __shared__ int ri[256];
    __shared__ float topv[KSEL_];
    __shared__ int topi[KSEL_];
    const int tid = threadIdx.x;

    for (int i = tid; i < VARN_; i += 256) vals[i] = src[i];
    __syncthreads();

    for (int k = 0; k < KSEL_; ++k) {
        float bv = -3.4e38f;
        int bi = 0x7fffffff;
        for (int i = tid; i < VARN_; i += 256) {
            float v = vals[i];
            if (v > bv || (v == bv && i < bi)) { bv = v; bi = i; }
        }
        rv[tid] = bv; ri[tid] = bi;
        __syncthreads();
        for (int st = 128; st > 0; st >>= 1) {
            if (tid < st) {
                float ov = rv[tid + st]; int oi = ri[tid + st];
                if (ov > rv[tid] || (ov == rv[tid] && oi < ri[tid])) {
                    rv[tid] = ov; ri[tid] = oi;
                }
            }
            __syncthreads();
        }
        if (tid == 0) { topv[k] = rv[0]; topi[k] = ri[0]; vals[ri[0]] = -3.4e38f; }
        __syncthreads();
    }

    // softmax over the 87 (already sorted descending, max = topv[0])
    float mx = topv[0];
    float sacc = 0.f;
    for (int k = tid; k < KSEL_; k += 256) sacc += __expf(topv[k] - mx);
    rv[tid] = sacc;
    __syncthreads();
    for (int st = 128; st > 0; st >>= 1) {
        if (tid < st) rv[tid] += rv[tid + st];
        __syncthreads();
    }
    float inv = 1.f / rv[0];
    for (int k = tid; k < KSEL_; k += 256) {
        size_t o = (size_t)row * KSEL_ + k;
        attn_out[o] = __expf(topv[k] - mx) * inv;
        idx_out[o] = topi[k];
    }
}

// ---------------- K5: gather + grouped conv1d + minmax-exp, wave per (s,h,g) ------
__global__ __launch_bounds__(256) void gather_conv_kernel(const bf16* __restrict__ vb,
                                                          const int* __restrict__ idx,
                                                          const float* __restrict__ gw,
                                                          const float* __restrict__ gb,
                                                          bf16* __restrict__ o) {
    const int wg = blockIdx.x * 8 + (threadIdx.x >> 5);  // 0..30719 = (s*8+h)*10+g
    const int lane = threadIdx.x & 31;                   // d in [0,32)
    const int g = wg % GRP_;
    const int h = (wg / GRP_) % HDS_;
    const int s = wg / (GRP_ * HDS_);
    const int* irow = idx + (size_t)wg * KSEL_;

    float acc = gb[g];
    for (int k = 0; k < KSEL_; ++k) {
        int n = irow[k];
        float w = gw[g * KSEL_ + k];
        acc += w * (float)vb[((size_t)(s * VARN_ + n)) * DM_ + h * DH_ + lane];
    }
    // wave32 min/max over the 32 d_h lanes
    float mn = acc, mx = acc;
#pragma unroll
    for (int off = 16; off > 0; off >>= 1) {
        mn = fminf(mn, __shfl_xor(mn, off, 32));
        mx = fmaxf(mx, __shfl_xor(mx, off, 32));
    }
    float denom = fmaxf(mx - mn, 1e-6f);
    float r = __expf((acc - mn) / denom);
    // o layout: [s][g][h*32 + lane]  (bs, groups, d_model)
    o[((size_t)s * GRP_ + g) * DM_ + h * DH_ + lane] = (bf16)r;
}

// ---------------- host-side orchestration ----------------
extern "C" void kernel_launch(void* const* d_in, const int* in_sizes, int n_in,
                              void* d_out, int out_size, void* d_ws, size_t ws_size,
                              hipStream_t stream) {
    (void)in_sizes; (void)n_in; (void)out_size; (void)ws_size;
    const float* x  = (const float*)d_in[0];
    const float* cl = (const float*)d_in[1];
    const float* Wq = (const float*)d_in[2];
    const float* bq = (const float*)d_in[3];
    const float* Wk = (const float*)d_in[4];
    const float* bk = (const float*)d_in[5];
    const float* Wv = (const float*)d_in[6];
    const float* bv = (const float*)d_in[7];
    const float* gw = (const float*)d_in[8];
    const float* gb = (const float*)d_in[9];
    const float* Wo = (const float*)d_in[10];
    const float* bo = (const float*)d_in[11];

    // workspace carve (all region sizes are multiples of 256B)
    char* ws = (char*)d_ws;
    size_t off = 0;
    auto carve = [&](size_t bytes) {
        void* p = ws + off;
        off += (bytes + 255) & ~(size_t)255;
        return p;
    };
    bf16*  yb   = (bf16*)carve((size_t)M1_ * DM_ * 2);                    // activations
    bf16*  kb   = (bf16*)carve((size_t)M1_ * DM_ * 2);                    // k
    bf16*  vb   = (bf16*)carve((size_t)M1_ * DM_ * 2);                    // v
    float* sc   = (float*)carve((size_t)BS_ * HDS_ * GRP_ * VARN_ * 4);   // scores
    bf16*  qb   = (bf16*)carve((size_t)SEGN_ * HDS_ * GRP_ * DH_ * 2);    // q (bf16)
    bf16*  WkT  = (bf16*)carve((size_t)DM_ * DM_ * 2);                    // transposed
    bf16*  WvT  = (bf16*)carve((size_t)DM_ * DM_ * 2);
    bf16*  WoT  = (bf16*)carve((size_t)DM_ * DM_ * 2);
    bf16*  ob   = (bf16*)carve((size_t)BS_ * GRP_ * DM_ * 2);             // o (pre-Wo)

    // d_out: [out | attn_k | idx]
    float* outF   = (float*)d_out;                                   // 384*10*256
    float* attnF  = outF + (size_t)BS_ * GRP_ * DM_;                 // 384*8*10*87
    int*   idxI   = (int*)d_out + (size_t)BS_ * GRP_ * DM_ +
                    (size_t)BS_ * HDS_ * GRP_ * KSEL_;

    // K0: weight casts (transposed for async row-staging) + q
    cast_wT_kernel<<<256, 256, 0, stream>>>(Wk, WkT);
    cast_wT_kernel<<<256, 256, 0, stream>>>(Wv, WvT);
    cast_wT_kernel<<<256, 256, 0, stream>>>(Wo, WoT);
    compute_q_kernel<<<240, 256, 0, stream>>>(cl, Wq, bq, qb);

    // K1: activation + bf16 cast
    {
        int nv = (M1_ * DM_) / 4;                 // 21,184,512
        act_cast_kernel<<<nv / 256, 256, 0, stream>>>((const float4*)x, (v4bf*)yb, nv);
    }

    // K2: k = y@Wk+bk and v = y@Wv+bv fused over gridDim.z (bf16 out)
    gemm_bf16_256<false><<<dim3(M1_ / 128, 2, 2), 256, 0, stream>>>(
        yb, WkT, WvT, bk, bv, kb, vb);

    // K3: scores
    scores_kernel<<<BS_ * HDS_, 256, 0, stream>>>(qb, kb, sc);

    // K4: top-k + softmax -> attn_k, idx (directly into d_out)
    topk_kernel<<<BS_ * HDS_ * GRP_, 256, 0, stream>>>(sc, attnF, idxI);

    // K5: gather + grouped conv + minmax-exp -> o (bf16)
    gather_conv_kernel<<<(BS_ * HDS_ * GRP_) / 8, 256, 0, stream>>>(vb, idxI, gw, gb, ob);

    // K6: out = o @ Wo + bo (f32 out into d_out)
    gemm_bf16_256<true><<<dim3((BS_ * GRP_) / 128, 2, 1), 256, 0, stream>>>(
        ob, WoT, WoT, bo, bo, outF, outF);
}
